// GraphQNNHybrid_65481071407851
// MI455X (gfx1250) — matrix operand order repaired and checked by reading, not compiled, not run on hardware
//
#include <hip/hip_runtime.h>
#include <hip/hip_bf16.h>
#include <math.h>

typedef __attribute__((ext_vector_type(16))) _Float16 v16h;
typedef __attribute__((ext_vector_type(8)))  _Float16 v8h;
typedef __attribute__((ext_vector_type(8)))  float    v8f;

#define FEAT 256

// ---------------------------------------------------------------------------
// Small utility kernels
// ---------------------------------------------------------------------------
__global__ void k_zero(float* __restrict__ p, int n) {
    int i = blockIdx.x * blockDim.x + threadIdx.x;
    if (i < n) p[i] = 0.0f;
}

__global__ void k_count(const int* __restrict__ src, float* __restrict__ cnt, int e) {
    int i = blockIdx.x * blockDim.x + threadIdx.x;
    if (i < e) atomicAdd(&cnt[src[i]], 1.0f);
}

__global__ void k_scatter(const int* __restrict__ src, const int* __restrict__ dst,
                          const float* __restrict__ rm, float* __restrict__ sums, int e) {
    int i = blockIdx.x * blockDim.x + threadIdx.x;
    if (i < e) atomicAdd(&sums[src[i]], rm[dst[i]]);
}

__global__ void k_nm(const float* __restrict__ sums, const float* __restrict__ cnt,
                     float* __restrict__ nm, int n) {
    int i = blockIdx.x * blockDim.x + threadIdx.x;
    if (i < n) {
        float c = cnt[i];
        nm[i] = (c > 0.0f) ? (sums[i] / fmaxf(c, 1.0f)) : 0.0f;
    }
}

// ---------------------------------------------------------------------------
// Fused WMMA GEMM:  gemmOut = f(gA) @ W + bias,  rowmean = mean(gemmOut, axis=1)
//   MODE 0: f(x) = x              (sideOut gets a copy of gA -> acts[0])
//   MODE 1: f(x) = tanh(x + nm)   (sideOut gets tanh result   -> acts[2])
// Block: 256 threads (8 waves). Tile: 128 rows x 256 cols, K staged 32 deep.
// Wave w owns rows [w*16, w*16+16) of the tile, all 16 column tiles.
// ---------------------------------------------------------------------------
template <int MODE>
__global__ __launch_bounds__(256)
void k_gemm(const float* __restrict__ gA, const float* __restrict__ nmIn,
            const float* __restrict__ W,  const float* __restrict__ bias,
            float* __restrict__ sideOut,  float* __restrict__ gemmOut,
            float* __restrict__ rowmean,  int n)
{
    __shared__ _Float16 As[128][40];   // 128x32 + pad
    __shared__ _Float16 Bs[32][264];   // 32x256 + pad

    const int tid  = threadIdx.x;
    const int lane = tid & 31;
    const int wave = tid >> 5;
    const int rowBase = blockIdx.x * 128;

    v8f acc[16];
    #pragma unroll
    for (int t = 0; t < 16; ++t)
        #pragma unroll
        for (int j = 0; j < 8; ++j) acc[t][j] = 0.0f;

    for (int kk = 0; kk < 8; ++kk) {
        // ---- stage A tile: 128 rows x 32 cols, fp32 -> f16 (+ fused producer) ----
        {
            const int ar = tid >> 1;            // 0..127
            const int ac = (tid & 1) * 16;      // 0 or 16
            const int gRow = rowBase + ar;
            const int gCol = kk * 32 + ac;
            float x[16];
            if (gRow < n) {
                const float4* p = (const float4*)(gA + (size_t)gRow * FEAT + gCol);
                float4 f0 = p[0], f1 = p[1], f2 = p[2], f3 = p[3];
                x[0]=f0.x; x[1]=f0.y; x[2]=f0.z;  x[3]=f0.w;
                x[4]=f1.x; x[5]=f1.y; x[6]=f1.z;  x[7]=f1.w;
                x[8]=f2.x; x[9]=f2.y; x[10]=f2.z; x[11]=f2.w;
                x[12]=f3.x; x[13]=f3.y; x[14]=f3.z; x[15]=f3.w;
                if (MODE == 1) {
                    float add = nmIn[gRow];
                    #pragma unroll
                    for (int i = 0; i < 16; ++i) x[i] = tanhf(x[i] + add);
                }
                float4* q = (float4*)(sideOut + (size_t)gRow * FEAT + gCol);
                q[0] = make_float4(x[0],  x[1],  x[2],  x[3]);
                q[1] = make_float4(x[4],  x[5],  x[6],  x[7]);
                q[2] = make_float4(x[8],  x[9],  x[10], x[11]);
                q[3] = make_float4(x[12], x[13], x[14], x[15]);
            } else {
                #pragma unroll
                for (int i = 0; i < 16; ++i) x[i] = 0.0f;
            }
            v8h h0, h1;
            #pragma unroll
            for (int i = 0; i < 8; ++i) { h0[i] = (_Float16)x[i]; h1[i] = (_Float16)x[8 + i]; }
            *(v8h*)&As[ar][ac]     = h0;
            *(v8h*)&As[ar][ac + 8] = h1;
        }
        // ---- stage B tile: 32 rows (K) x 256 cols, fp32 -> f16 ----
        {
            const int br = tid >> 3;            // 0..31
            const int bc = (tid & 7) * 32;      // 0..224
            const float4* p = (const float4*)(W + (size_t)(kk * 32 + br) * FEAT + bc);
            #pragma unroll
            for (int u = 0; u < 4; ++u) {
                float4 fa = p[2 * u], fb = p[2 * u + 1];
                v8h h;
                h[0]=(_Float16)fa.x; h[1]=(_Float16)fa.y; h[2]=(_Float16)fa.z; h[3]=(_Float16)fa.w;
                h[4]=(_Float16)fb.x; h[5]=(_Float16)fb.y; h[6]=(_Float16)fb.z; h[7]=(_Float16)fb.w;
                *(v8h*)&Bs[br][bc + 8 * u] = h;
            }
        }
        __syncthreads();

        // ---- compute: A fragment once, 16 WMMAs across column tiles ----
        {
            // A layout (16-bit 16x32): lanes 0-15 row m, K {0..7,16..23};
            //                          lanes 16-31 row m, K {8..15,24..31}
            const int m    = wave * 16 + (lane & 15);
            const int koff = (lane & 16) ? 8 : 0;
            v8h alo = *(const v8h*)&As[m][koff];
            v8h ahi = *(const v8h*)&As[m][koff + 16];
            v16h a;
            #pragma unroll
            for (int i = 0; i < 8; ++i) { a[i] = alo[i]; a[8 + i] = ahi[i]; }

            #pragma unroll
            for (int t = 0; t < 16; ++t) {
                // B layout (32x16): lane%16 = N column, K same as A: {koff..koff+7, koff+16..koff+23}
                const int bn = t * 16 + (lane & 15);
                v16h b;
                #pragma unroll
                for (int i = 0; i < 8; ++i) { b[i] = Bs[koff + i][bn]; b[8 + i] = Bs[koff + 16 + i][bn]; }
                acc[t] = __builtin_amdgcn_wmma_f32_16x16x32_f16(
                    false, a, false, b, (short)0, acc[t], false, false);
            }
        }
        __syncthreads();
    }

    // ---- epilogue: bias add, store, fused row-mean reduction ----
    // C layout: acc[t][r] == C[m = r + (lane&16 ? 8 : 0)][ncol = t*16 + (lane&15)]
    const int rShift = (lane & 16) ? 8 : 0;
    const int n0 = lane & 15;
    float rs[8];
    #pragma unroll
    for (int r = 0; r < 8; ++r) rs[r] = 0.0f;

    #pragma unroll
    for (int t = 0; t < 16; ++t) {
        const int ncol = t * 16 + n0;
        const float bv = bias[ncol];
        #pragma unroll
        for (int r = 0; r < 8; ++r) {
            const int gRow = rowBase + wave * 16 + rShift + r;
            float v = acc[t][r] + bv;
            rs[r] += v;
            if (gRow < n) gemmOut[(size_t)gRow * FEAT + ncol] = v;
        }
    }
    // reduce across the 16 lanes that share the same row set
    #pragma unroll
    for (int mask = 1; mask < 16; mask <<= 1)
        #pragma unroll
        for (int r = 0; r < 8; ++r) rs[r] += __shfl_xor(rs[r], mask, 32);
    if (n0 == 0) {
        #pragma unroll
        for (int r = 0; r < 8; ++r) {
            const int gRow = rowBase + wave * 16 + rShift + r;
            if (gRow < n) rowmean[gRow] = rs[r] * (1.0f / 256.0f);
        }
    }
}

// ---------------------------------------------------------------------------
// Final layer: acts4 = tanh(acts3 + nm2), acts5 = acts4 @ W3 + b3
// One row per wave; 8 cols per lane; shfl reduction for the 256-dot.
// ---------------------------------------------------------------------------
__global__ __launch_bounds__(256)
void k_final(const float* __restrict__ a3, const float* __restrict__ nm2,
             const float* __restrict__ W3, const float* __restrict__ b3,
             float* __restrict__ a4, float* __restrict__ a5, int n)
{
    const int lane = threadIdx.x & 31;
    const int wave = threadIdx.x >> 5;
    const int row = blockIdx.x * 8 + wave;
    if (row >= n) return;
    const float add = nm2[row];
    const float4* p  = (const float4*)(a3 + (size_t)row * FEAT + lane * 8);
    const float4* wv = (const float4*)(W3 + lane * 8);
    float4* q = (float4*)(a4 + (size_t)row * FEAT + lane * 8);
    float s = 0.0f;
    #pragma unroll
    for (int u = 0; u < 2; ++u) {
        float4 f = p[u], w = wv[u];
        float t0 = tanhf(f.x + add), t1 = tanhf(f.y + add);
        float t2 = tanhf(f.z + add), t3 = tanhf(f.w + add);
        q[u] = make_float4(t0, t1, t2, t3);
        s += t0 * w.x + t1 * w.y + t2 * w.z + t3 * w.w;
    }
    #pragma unroll
    for (int mask = 1; mask < 32; mask <<= 1) s += __shfl_xor(s, mask, 32);
    if (lane == 0) a5[row] = s + b3[0];
}

// ---------------------------------------------------------------------------
extern "C" void kernel_launch(void* const* d_in, const int* in_sizes, int n_in,
                              void* d_out, int out_size, void* d_ws, size_t ws_size,
                              hipStream_t stream)
{
    const float* data = (const float*)d_in[0];
    const int*   esrc = (const int*)d_in[1];
    const int*   edst = (const int*)d_in[2];
    const float* W1   = (const float*)d_in[3];
    const float* b1   = (const float*)d_in[4];
    const float* W2   = (const float*)d_in[5];
    const float* b2   = (const float*)d_in[6];
    const float* W3   = (const float*)d_in[7];
    const float* b3   = (const float*)d_in[8];

    const int n = in_sizes[0] / FEAT;   // 100000 nodes
    const int e = in_sizes[1];          // 3200000 edges

    float* out = (float*)d_out;
    const size_t plane = (size_t)n * FEAT;
    float* acts0 = out;
    float* acts1 = out + plane;
    float* acts2 = out + 2 * plane;
    float* acts3 = out + 3 * plane;
    float* acts4 = out + 4 * plane;
    float* acts5 = out + 5 * plane;

    float* ws   = (float*)d_ws;
    float* rm   = ws;                       // row means (reused between rounds)
    float* cnt  = ws + (size_t)n;           // degree counts (computed once)
    float* sums = ws + 2 * (size_t)n;       // segment sums
    float* nm1  = ws + 3 * (size_t)n;
    float* nm2  = ws + 4 * (size_t)n;

    const int tpb = 256;
    const int gN = (n + tpb - 1) / tpb;
    const int gE = (e + tpb - 1) / tpb;
    const int gG = (n + 127) / 128;
    const int gF = (n + 7) / 8;

    // zero cnt+sums (contiguous, 2N floats) fresh every launch
    k_zero<<<(2 * n + tpb - 1) / tpb, tpb, 0, stream>>>(cnt, 2 * n);

    // layer 1: acts0 = copy(data), acts1 = data@W1+b1, rm = rowmean(acts1)
    k_gemm<0><<<gG, tpb, 0, stream>>>(data, nullptr, W1, b1, acts0, acts1, rm, n);

    // neighbor mean round 1
    k_count  <<<gE, tpb, 0, stream>>>(esrc, cnt, e);
    k_scatter<<<gE, tpb, 0, stream>>>(esrc, edst, rm, sums, e);
    k_nm     <<<gN, tpb, 0, stream>>>(sums, cnt, nm1, n);

    // layer 2: acts2 = tanh(acts1+nm1), acts3 = acts2@W2+b2, rm = rowmean(acts3)
    k_gemm<1><<<gG, tpb, 0, stream>>>(acts1, nm1, W2, b2, acts2, acts3, rm, n);

    // neighbor mean round 2 (reuse cnt)
    k_zero   <<<gN, tpb, 0, stream>>>(sums, n);
    k_scatter<<<gE, tpb, 0, stream>>>(esrc, edst, rm, sums, e);
    k_nm     <<<gN, tpb, 0, stream>>>(sums, cnt, nm2, n);

    // layer 3: acts4 = tanh(acts3+nm2), acts5 = acts4@W3+b3
    k_final<<<gF, tpb, 0, stream>>>(acts3, nm2, W3, b3, acts4, acts5, n);
}
